// MegalodonAttention_25563645345980
// MI455X (gfx1250) — matrix operand order repaired and checked by reading, not compiled, not run on hardware
//
#include <hip/hip_runtime.h>
#include <hip/hip_bf16.h>

typedef __attribute__((ext_vector_type(16))) _Float16 v16h;
typedef __attribute__((ext_vector_type(8)))  _Float16 v8h;
typedef __attribute__((ext_vector_type(8)))  float    v8f;

#define B_     4
#define L_     4096
#define H_     16
#define DH_    128
#define DV_    128
#define CHUNK_ 1024
#define NC_    (L_ / CHUNK_)
#define BM     128          // query rows per block (8 waves x 16)
#define BN     64           // keys per inner block
#define NWAVE  8

// LDS strides in halves; rows must be multiples of 8 halves (16B) for b128 loads
#define KS_STRIDE 136       // DH + 8  -> 272B row
#define VT_STRIDE 72        // BN + 8  -> 144B row
#define PS_STRIDE 72

#define PS_HALVES (NWAVE * 16 * PS_STRIDE)   // 9216
#define VT_HALVES (DV_ * VT_STRIDE)          // 9216

__device__ __forceinline__ v16h cat8(v8h lo, v8h hi) {
  return __builtin_shufflevector(lo, hi, 0,1,2,3,4,5,6,7,8,9,10,11,12,13,14,15);
}

__global__ __launch_bounds__(256) void
megalodon_attn_gfx1250(const float* __restrict__ q,
                       const float* __restrict__ k,
                       const float* __restrict__ v,
                       float* __restrict__ out)
{
  constexpr float ROPE_C = 0.14391156537f;        // ln(10000)/64
  constexpr float QSCALE = 0.08838834764831845f;  // 1/sqrt(128)
  const float NEG_INF = -__builtin_inff();

  __shared__ __align__(32) _Float16 smem[PS_HALVES + VT_HALVES];
  _Float16* Ks = smem;                 // unioned with Ps (barrier-protected)
  _Float16* Ps = smem;
  _Float16* Vt = smem + PS_HALVES;

  const int tid  = threadIdx.x;
  const int wv   = tid >> 5;
  const int lane = tid & 31;
  const int lr   = lane & 15;
  const int hi   = lane >> 4;          // half-wave group

  int bid = blockIdx.x;
  const int h  = bid % H_;             bid /= H_;
  const int qt = bid % (CHUNK_ / BM);  bid /= (CHUNK_ / BM);
  const int n  = bid % NC_;            bid /= NC_;
  const int b  = bid;

  const int qbase = qt * BM;
  const int qrow0 = qbase + wv * 16;

  // ---- Q fragments: RoPE + scale folded, f16, contiguous A-layout ----
  // RoPE pairs (dh, dh+64) share one angle: compute frags {0,2} and {1,3} together.
  const int gq = n * CHUNK_ + qrow0 + lr;  // global seq position of this lane's row
  const float* qp = q + (((size_t)b * L_ + gq) * H_ + h) * DH_;
  v16h qfrag[4];
#pragma unroll
  for (int ks = 0; ks < 2; ++ks) {
#pragma unroll
    for (int e = 0; e < 16; ++e) {
      const int f = 32 * ks + hi * 16 + e;  // in [0,64)
      const float inv = __expf(-(float)f * ROPE_C);
      float sn, cs; __sincosf((float)gq * inv, &sn, &cs);
      const float x1 = qp[f], x2 = qp[f + 64];
      qfrag[ks]     [e] = (_Float16)((x1 * cs - x2 * sn) * QSCALE);
      qfrag[ks + 2][e] = (_Float16)((x1 * sn + x2 * cs) * QSCALE);
    }
  }

  // ---- loop-invariant staging constants ----
  const int fK = tid & 63;             // K-staging frequency index (constant)
  const int j0 = tid >> 6;             // first staged key (0..3), step 4
  const float invK = __expf(-(float)fK * ROPE_C);
  float s4, c4; __sincosf(4.0f * invK, &s4, &c4);   // rotation per j += 4
  const int dvV = tid & 127;           // V-staging column (constant)
  const int jV0 = tid >> 7;            // first staged V key (0..1), step 2

  v8f oacc[8];
#pragma unroll
  for (int ot = 0; ot < 8; ++ot) oacc[ot] = {};
  float mrun[8], lrun[8];
#pragma unroll
  for (int i = 0; i < 8; ++i) { mrun[i] = NEG_INF; lrun[i] = 0.f; }

  const int kend = qbase + BM;  // causal upper bound for this q-tile
  for (int kb0 = 0; kb0 < kend; kb0 += BN) {
    __syncthreads();  // prior iteration's Ps/Vt consumers are done

    // ---- stage K (RoPE'd, f16, row-major); one sincos + Givens recurrence ----
    {
      const int tk0 = n * CHUNK_ + kb0 + j0;
      float sn, cs; __sincosf((float)tk0 * invK, &sn, &cs);
      const float* kp = k + (((size_t)b * L_ + tk0) * H_ + h) * DH_;
#pragma unroll
      for (int i = 0; i < 16; ++i) {
        const int j = j0 + 4 * i;
        const float x1 = kp[fK], x2 = kp[fK + 64];
        Ks[j * KS_STRIDE + fK]      = (_Float16)(x1 * cs - x2 * sn);
        Ks[j * KS_STRIDE + fK + 64] = (_Float16)(x1 * sn + x2 * cs);
        const float csn = cs * c4 - sn * s4;   // advance angle by 4*invK
        sn = sn * c4 + cs * s4;
        cs = csn;
        kp += 4 * H_ * DH_;
      }
    }
    // ---- stage V transposed: Vt[dv][key] ----
    {
      const float* vp = v + (((size_t)b * L_ + n * CHUNK_ + kb0 + jV0) * H_ + h) * DV_ + dvV;
#pragma unroll
      for (int i = 0; i < 32; ++i) {
        Vt[dvV * VT_STRIDE + jV0 + 2 * i] = (_Float16)(*vp);
        vp += 2 * H_ * DV_;
      }
    }
    // prefetch next K block into GL2
    if (kb0 + BN < kend) {
      const int tkn = n * CHUNK_ + kb0 + BN + (tid & 63);
      __builtin_prefetch(k + (((size_t)b * L_ + tkn) * H_ + h) * DH_, 0, 0);
    }
    __syncthreads();

    // ---- S = Q * K^T : A operand held across the inner 4 WMMAs ----
    v8f s[4];
#pragma unroll
    for (int t = 0; t < 4; ++t) s[t] = {};
#pragma unroll
    for (int ks = 0; ks < 4; ++ks) {
#pragma unroll
      for (int t = 0; t < 4; ++t) {
        const int j = t * 16 + lr;  // key column owned by this lane (B-layout)
        v16h bf = *(const v16h*)&Ks[j * KS_STRIDE + 32 * ks + hi * 16];
        s[t] = __builtin_amdgcn_wmma_f32_16x16x32_f16(
            false, qfrag[ks], false, bf, (short)0, s[t], false, false);
      }
    }
    __syncthreads();  // all waves done reading Ks before Ps (union) is written

    // ---- causal mask (C/D layout: row = i + 8*hi, col = lr) ----
    if (kb0 + BN - 1 > qrow0) {
#pragma unroll
      for (int t = 0; t < 4; ++t) {
        const int ki = kb0 + t * 16 + lr;
#pragma unroll
        for (int i = 0; i < 8; ++i) {
          const int qi = qrow0 + i + 8 * hi;
          if (ki > qi) s[t][i] = NEG_INF;
        }
      }
    }

    // ---- online softmax (rows 0-7 in lanes 0-15, rows 8-15 in lanes 16-31) ----
    float bmax[8];
#pragma unroll
    for (int i = 0; i < 8; ++i)
      bmax[i] = fmaxf(fmaxf(s[0][i], s[1][i]), fmaxf(s[2][i], s[3][i]));
#pragma unroll
    for (int d = 1; d <= 8; d <<= 1)
#pragma unroll
      for (int i = 0; i < 8; ++i)
        bmax[i] = fmaxf(bmax[i], __shfl_xor(bmax[i], d, 32));

    float alpha[8];
#pragma unroll
    for (int i = 0; i < 8; ++i) {
      const float mn = fmaxf(mrun[i], bmax[i]);
      alpha[i] = __expf(mrun[i] - mn);
      mrun[i] = mn;
    }
#pragma unroll
    for (int t = 0; t < 4; ++t)
#pragma unroll
      for (int i = 0; i < 8; ++i)
        s[t][i] = __expf(s[t][i] - mrun[i]);

    float rsum[8];
#pragma unroll
    for (int i = 0; i < 8; ++i)
      rsum[i] = (s[0][i] + s[1][i]) + (s[2][i] + s[3][i]);
#pragma unroll
    for (int d = 1; d <= 8; d <<= 1)
#pragma unroll
      for (int i = 0; i < 8; ++i)
        rsum[i] += __shfl_xor(rsum[i], d, 32);
#pragma unroll
    for (int i = 0; i < 8; ++i)
      lrun[i] = lrun[i] * alpha[i] + rsum[i];

#pragma unroll
    for (int ot = 0; ot < 8; ++ot)
#pragma unroll
      for (int i = 0; i < 8; ++i)
        oacc[ot][i] *= alpha[i];

    // ---- P (D-layout) -> LDS -> A-layout fragments; O += P * V ----
    _Float16* Pw = Ps + wv * 16 * PS_STRIDE;
#pragma unroll
    for (int t = 0; t < 4; ++t)
#pragma unroll
      for (int i = 0; i < 8; ++i)
        Pw[(i + 8 * hi) * PS_STRIDE + t * 16 + lr] = (_Float16)s[t][i];
    // same-wave LDS store->load is in-order; compiler inserts s_wait_dscnt

#pragma unroll
    for (int ks2 = 0; ks2 < 2; ++ks2) {
      const int ob = 32 * ks2 + hi * 16;
      v8h plo = *(const v8h*)&Pw[lr * PS_STRIDE + ob];
      v8h phi = *(const v8h*)&Pw[lr * PS_STRIDE + ob + 8];
      v16h pa = cat8(plo, phi);   // A operand held across the 8 WMMAs below
#pragma unroll
      for (int ot = 0; ot < 8; ++ot) {
        const int dv = ot * 16 + lr;
        v16h vb = *(const v16h*)&Vt[dv * VT_STRIDE + 32 * ks2 + hi * 16];
        oacc[ot] = __builtin_amdgcn_wmma_f32_16x16x32_f16(
            false, pa, false, vb, (short)0, oacc[ot], false, false);
      }
    }
  }

  // ---- normalize and store (out[b, l, h*DV + dv], fp32) ----
  float invl[8];
#pragma unroll
  for (int i = 0; i < 8; ++i) invl[i] = 1.0f / lrun[i];
#pragma unroll
  for (int ot = 0; ot < 8; ++ot) {
#pragma unroll
    for (int i = 0; i < 8; ++i) {
      const int m  = i + 8 * hi;
      const int gl = n * CHUNK_ + qbase + wv * 16 + m;
      out[((size_t)b * L_ + gl) * (H_ * DV_) + h * DV_ + ot * 16 + lr] =
          oacc[ot][i] * invl[i];
    }
  }
}

extern "C" void kernel_launch(void* const* d_in, const int* in_sizes, int n_in,
                              void* d_out, int out_size, void* d_ws, size_t ws_size,
                              hipStream_t stream) {
  (void)in_sizes; (void)n_in; (void)out_size; (void)d_ws; (void)ws_size;
  const float* q = (const float*)d_in[0];
  const float* k = (const float*)d_in[1];
  const float* v = (const float*)d_in[2];
  float* out = (float*)d_out;
  // (b, chunk, q-tile, head) -> 4*4*8*16 = 2048 workgroups of 256 threads
  dim3 grid(B_ * NC_ * (CHUNK_ / BM) * H_);
  megalodon_attn_gfx1250<<<grid, 256, 0, stream>>>(q, k, v, out);
}
